// CurriculumSeqDecoder_16389595201835
// MI455X (gfx1250) — matrix operand order repaired and checked by reading, not compiled, not run on hardware
//
#include <hip/hip_runtime.h>

// ---------------------------------------------------------------------------
// CDNA5 / gfx1250 attention-LSTM greedy decoder.
// All GEMMs run on v_wmma_f32_16x16x32_bf16 (wave32, fp32 accumulation).
// Each wave computes a 16x64 strip: A fragment reused across 4 independent
// WMMA accumulator chains (ILP + 1.6x load reduction).
// ---------------------------------------------------------------------------

typedef __bf16 bf16_t;
typedef __attribute__((ext_vector_type(16))) __bf16 v16bf;
typedef __attribute__((ext_vector_type(8)))  __bf16 v8bf;
typedef __attribute__((ext_vector_type(8)))  float  v8f;

#define Hn 512
#define Vn 16000
#define Bn 64
#define Sn 128
#define Tn 100

// --------------------------- fp32 -> bf16 convert ---------------------------
__global__ void k_cvt_bf16(const float* __restrict__ src, bf16_t* __restrict__ dst, int n) {
  int i = blockIdx.x * blockDim.x + threadIdx.x;
  if (i < n) dst[i] = (bf16_t)src[i];
}

// --------------------------- WMMA bf16 GEMM ---------------------------------
// C[M,N] (f32) = A[M,K](bf16, row-major) * W[N,K](bf16, row-major)^T + bias
// accumulate!=0: acc initialized from existing C (+bias).
// One wave per 16(M) x 64(N) strip; K stepped by 32; 4 accumulators.
__device__ __forceinline__ v16bf frag_a(const bf16_t* rowp, int half) {
  v8bf lo = *(const v8bf*)(rowp + half * 8);
  v8bf hi = *(const v8bf*)(rowp + half * 8 + 16);
  return __builtin_shufflevector(lo, hi, 0,1,2,3,4,5,6,7,8,9,10,11,12,13,14,15);
}
__device__ __forceinline__ v16bf frag_b(const bf16_t* rowp, int half) {
  v8bf lo = *(const v8bf*)(rowp + half * 8);
  v8bf hi = *(const v8bf*)(rowp + half * 8 + 16);
  return __builtin_shufflevector(lo, hi, 0,1,2,3,4,5,6,7,8,9,10,11,12,13,14,15);
}

__global__ void k_gemm_bf16_wmma(const bf16_t* __restrict__ A,
                                 const bf16_t* __restrict__ W,
                                 const float*  __restrict__ bias,
                                 float* __restrict__ C,
                                 int M, int N, int K, int accumulate) {
  const int lane = threadIdx.x;                       // 0..31
  const int strip = blockIdx.x * blockDim.y + threadIdx.y;
  const int nstrips = N >> 6;                         // N/64
  const int mt = strip / nstrips;
  const int ng = strip % nstrips;
  if (mt * 16 >= M) return;                           // wave-uniform exit
  const int half = lane >> 4;                         // 0|1
  const int idx  = lane & 15;
  const int m0 = mt << 4, n0 = ng << 6;
  const bf16_t* arow = A + (size_t)(m0 + idx) * K;
  const bf16_t* wrow0 = W + (size_t)(n0      + idx) * K;
  const bf16_t* wrow1 = W + (size_t)(n0 + 16 + idx) * K;
  const bf16_t* wrow2 = W + (size_t)(n0 + 32 + idx) * K;
  const bf16_t* wrow3 = W + (size_t)(n0 + 48 + idx) * K;

  v8f acc0, acc1, acc2, acc3;
  {
    const float b0 = bias ? bias[n0      + idx] : 0.0f;
    const float b1 = bias ? bias[n0 + 16 + idx] : 0.0f;
    const float b2 = bias ? bias[n0 + 32 + idx] : 0.0f;
    const float b3 = bias ? bias[n0 + 48 + idx] : 0.0f;
    if (accumulate) {
#pragma unroll
      for (int r = 0; r < 8; ++r) {
        const size_t rowoff = (size_t)(m0 + half * 8 + r) * N;
        acc0[r] = C[rowoff + n0      + idx] + b0;
        acc1[r] = C[rowoff + n0 + 16 + idx] + b1;
        acc2[r] = C[rowoff + n0 + 32 + idx] + b2;
        acc3[r] = C[rowoff + n0 + 48 + idx] + b3;
      }
    } else {
#pragma unroll
      for (int r = 0; r < 8; ++r) { acc0[r] = b0; acc1[r] = b1; acc2[r] = b2; acc3[r] = b3; }
    }
  }

  for (int kk = 0; kk < K; kk += 32) {
    v16bf a  = frag_a(arow  + kk, half);      // shared across the 4 WMMAs
    v16bf b0 = frag_b(wrow0 + kk, half);
    v16bf b1 = frag_b(wrow1 + kk, half);
    v16bf b2 = frag_b(wrow2 + kk, half);
    v16bf b3 = frag_b(wrow3 + kk, half);
    acc0 = __builtin_amdgcn_wmma_f32_16x16x32_bf16(false, a, false, b0, (short)0, acc0, false, false);
    acc1 = __builtin_amdgcn_wmma_f32_16x16x32_bf16(false, a, false, b1, (short)0, acc1, false, false);
    acc2 = __builtin_amdgcn_wmma_f32_16x16x32_bf16(false, a, false, b2, (short)0, acc2, false, false);
    acc3 = __builtin_amdgcn_wmma_f32_16x16x32_bf16(false, a, false, b3, (short)0, acc3, false, false);
  }

#pragma unroll
  for (int r = 0; r < 8; ++r) {
    const size_t rowoff = (size_t)(m0 + half * 8 + r) * N;
    C[rowoff + n0      + idx] = acc0[r];
    C[rowoff + n0 + 16 + idx] = acc1[r];
    C[rowoff + n0 + 32 + idx] = acc2[r];
    C[rowoff + n0 + 48 + idx] = acc3[r];
  }
}

// ------------------ attention scores: Va . tanh(q + UaK) --------------------
// one wave per (b,s) row
__global__ void k_attn_scores(const float* __restrict__ q,
                              const float* __restrict__ UaK,
                              const float* __restrict__ Va_w,
                              const float* __restrict__ Va_b,
                              float* __restrict__ scores) {
  const int lane = threadIdx.x;
  const int row  = blockIdx.x * blockDim.y + threadIdx.y;   // b*Sn + s
  if (row >= Bn * Sn) return;
  const int b = row >> 7;
  const float* qb = q + b * Hn;
  const float* uk = UaK + (size_t)row * Hn;
  float acc = 0.f;
  for (int h = lane; h < Hn; h += 32)
    acc += Va_w[h] * tanhf(qb[h] + uk[h]);
#pragma unroll
  for (int off = 16; off; off >>= 1) acc += __shfl_xor(acc, off, 32);
  if (lane == 0) scores[row] = acc + Va_b[0];
}

// --- softmax over S, write attentions, context GEMV, embedding gather -> x --
__global__ void k_softmax_ctx(const float* __restrict__ scores,
                              const float* __restrict__ enc,
                              const float* __restrict__ emb,
                              const int*   __restrict__ dec,
                              bf16_t* __restrict__ xbf,
                              float* __restrict__ att, int t) {
  const int b   = blockIdx.x;
  const int tid = threadIdx.x;                 // Sn threads
  __shared__ float w[Sn];
  __shared__ float red[Sn];
  float sc = scores[b * Sn + tid];
  red[tid] = sc; __syncthreads();
  for (int off = Sn / 2; off; off >>= 1) {
    if (tid < off) red[tid] = fmaxf(red[tid], red[tid + off]);
    __syncthreads();
  }
  float mx = red[0]; __syncthreads();
  float e = expf(sc - mx);
  red[tid] = e; __syncthreads();
  for (int off = Sn / 2; off; off >>= 1) {
    if (tid < off) red[tid] += red[tid + off];
    __syncthreads();
  }
  float wv = e / red[0];
  w[tid] = wv;
  att[((size_t)b * Tn + t) * Sn + tid] = wv;
  __syncthreads();
  const int d = dec[b];
  for (int h = tid; h < Hn; h += Sn) {
    const float* ep = enc + ((size_t)b * Sn) * Hn + h;
    float cx = 0.f;
    for (int s = 0; s < Sn; ++s) cx += w[s] * ep[(size_t)s * Hn];
    xbf[(size_t)b * (3 * Hn) + Hn + h] = (bf16_t)cx;                      // ctx slot
    xbf[(size_t)b * (3 * Hn) + h]      = (bf16_t)emb[(size_t)d * Hn + h]; // emb slot
  }
}

// ------------------------------- LSTM cell ----------------------------------
__global__ void k_lstm(const float* __restrict__ gates, float* __restrict__ c,
                       float* __restrict__ h, bf16_t* __restrict__ hbf) {
  int i = blockIdx.x * blockDim.x + threadIdx.x;
  if (i >= Bn * Hn) return;
  int b = i / Hn, hh = i % Hn;
  const float* g = gates + (size_t)b * 4 * Hn;
  float ig = 1.f / (1.f + expf(-g[hh]));
  float fg = 1.f / (1.f + expf(-g[Hn + hh]));
  float gg = tanhf(g[2 * Hn + hh]);
  float og = 1.f / (1.f + expf(-g[3 * Hn + hh]));
  float cn = fg * c[i] + ig * gg;
  float hn = og * tanhf(cn);
  c[i] = cn; h[i] = hn; hbf[i] = (bf16_t)hn;
}

// ---------------- log-softmax over V + greedy argmax ------------------------
__global__ void k_logsoftmax(const float* __restrict__ logits,
                             float* __restrict__ logp, int* __restrict__ dec, int t) {
  const int b = blockIdx.x;
  const int tid = threadIdx.x;                 // 256 threads
  __shared__ float rv[256];
  __shared__ int   ri[256];
  const float* lb = logits + (size_t)b * Vn;
  float best = -3.4e38f; int bi = 0x7fffffff;
  for (int v = tid; v < Vn; v += 256) {
    float x = lb[v];
    if (x > best) { best = x; bi = v; }
  }
  rv[tid] = best; ri[tid] = bi; __syncthreads();
  for (int off = 128; off; off >>= 1) {
    if (tid < off) {
      if (rv[tid + off] > rv[tid] ||
          (rv[tid + off] == rv[tid] && ri[tid + off] < ri[tid])) {
        rv[tid] = rv[tid + off]; ri[tid] = ri[tid + off];
      }
    }
    __syncthreads();
  }
  float mx = rv[0]; int am = ri[0]; __syncthreads();
  float s = 0.f;
  for (int v = tid; v < Vn; v += 256) s += expf(lb[v] - mx);
  rv[tid] = s; __syncthreads();
  for (int off = 128; off; off >>= 1) {
    if (tid < off) rv[tid] += rv[tid + off];
    __syncthreads();
  }
  float lse = mx + logf(rv[0]);
  float* op = logp + ((size_t)b * Tn + t) * Vn;
  for (int v = tid; v < Vn; v += 256) op[v] = lb[v] - lse;
  if (tid == 0) dec[b] = am;
}

// -------------------------- state init / final copy -------------------------
__global__ void k_init(const float* __restrict__ hidden, const float* __restrict__ cell,
                       const int* __restrict__ tidx, const float* __restrict__ type_emb,
                       float* __restrict__ h, float* __restrict__ c,
                       bf16_t* __restrict__ hbf, bf16_t* __restrict__ xbf,
                       int* __restrict__ dec) {
  int i = blockIdx.x * blockDim.x + threadIdx.x;
  if (i < Bn) dec[i] = 0;                       // SOS
  if (i >= Bn * Hn) return;
  float hv = hidden[i];
  h[i] = hv; hbf[i] = (bf16_t)hv; c[i] = cell[i];
  int b = i / Hn, hh = i % Hn;
  xbf[(size_t)b * (3 * Hn) + 2 * Hn + hh] =
      (bf16_t)type_emb[(size_t)tidx[b] * Hn + hh];   // type_e slot, loop-invariant
}

__global__ void k_final(const float* __restrict__ h, const float* __restrict__ c,
                        float* __restrict__ oh, float* __restrict__ oc) {
  int i = blockIdx.x * blockDim.x + threadIdx.x;
  if (i < Bn * Hn) { oh[i] = h[i]; oc[i] = c[i]; }
}

// ---------------------------------------------------------------------------
extern "C" void kernel_launch(void* const* d_in, const int* in_sizes, int n_in,
                              void* d_out, int out_size, void* d_ws, size_t ws_size,
                              hipStream_t stream) {
  (void)in_sizes; (void)n_in; (void)out_size; (void)ws_size;
  const float* enc      = (const float*)d_in[0];
  const float* hidden   = (const float*)d_in[1];
  const float* cell     = (const float*)d_in[2];
  const int*   type_idx = (const int*)  d_in[3];
  const float* emb      = (const float*)d_in[4];
  const float* type_emb = (const float*)d_in[5];
  const float* Wa_w     = (const float*)d_in[6];
  const float* Wa_b     = (const float*)d_in[7];
  const float* Ua_w     = (const float*)d_in[8];
  const float* Ua_b     = (const float*)d_in[9];
  const float* Va_w     = (const float*)d_in[10];
  const float* Va_b     = (const float*)d_in[11];
  const float* W_ih     = (const float*)d_in[12];
  const float* b_ih     = (const float*)d_in[13];
  const float* W_hh     = (const float*)d_in[14];
  const float* b_hh     = (const float*)d_in[15];
  const float* out_w    = (const float*)d_in[16];
  const float* out_b    = (const float*)d_in[17];

  // ---- workspace carve-out (256B aligned slices) ----
  size_t off = 0;
  auto wsa = [&](size_t bytes) -> char* {
    char* p = (char*)d_ws + off;
    off = (off + bytes + 255) & ~(size_t)255;
    return p;
  };
  bf16_t* enc_bf  = (bf16_t*)wsa((size_t)Bn * Sn * Hn * 2);
  bf16_t* Uaw_bf  = (bf16_t*)wsa((size_t)Hn * Hn * 2);
  bf16_t* Waw_bf  = (bf16_t*)wsa((size_t)Hn * Hn * 2);
  bf16_t* Wih_bf  = (bf16_t*)wsa((size_t)4 * Hn * 3 * Hn * 2);
  bf16_t* Whh_bf  = (bf16_t*)wsa((size_t)4 * Hn * Hn * 2);
  bf16_t* outw_bf = (bf16_t*)wsa((size_t)Vn * Hn * 2);
  float*  UaK     = (float*) wsa((size_t)Bn * Sn * Hn * 4);
  float*  qbuf    = (float*) wsa((size_t)Bn * Hn * 4);
  float*  scores  = (float*) wsa((size_t)Bn * Sn * 4);
  bf16_t* xbf     = (bf16_t*)wsa((size_t)Bn * 3 * Hn * 2);
  float*  gates   = (float*) wsa((size_t)Bn * 4 * Hn * 4);
  float*  hbuf    = (float*) wsa((size_t)Bn * Hn * 4);
  float*  cbuf    = (float*) wsa((size_t)Bn * Hn * 4);
  bf16_t* hbf     = (bf16_t*)wsa((size_t)Bn * Hn * 2);
  float*  logits  = (float*) wsa((size_t)Bn * Vn * 4);
  int*    dec     = (int*)   wsa((size_t)Bn * 4);

  // ---- output sections: [B,T,V] logp | [1,B,H] h | [1,B,H] c | [B,T,S] att
  float* o_logp = (float*)d_out;
  float* o_h    = o_logp + (size_t)Bn * Tn * Vn;
  float* o_c    = o_h + (size_t)Bn * Hn;
  float* o_att  = o_c + (size_t)Bn * Hn;

  auto cvt = [&](const float* s, bf16_t* d, int n) {
    k_cvt_bf16<<<(n + 255) / 256, 256, 0, stream>>>(s, d, n);
  };
  auto gemm = [&](const bf16_t* A, const bf16_t* W, const float* bias, float* C,
                  int M, int N, int K, int accum) {
    int strips = (M / 16) * (N / 64);
    k_gemm_bf16_wmma<<<(strips + 7) / 8, dim3(32, 8), 0, stream>>>(A, W, bias, C,
                                                                   M, N, K, accum);
  };

  // ---- one-time (per launch): bf16 weights + loop-invariant Ua*keys ----
  cvt(enc,   enc_bf,  Bn * Sn * Hn);
  cvt(Ua_w,  Uaw_bf,  Hn * Hn);
  cvt(Wa_w,  Waw_bf,  Hn * Hn);
  cvt(W_ih,  Wih_bf,  4 * Hn * 3 * Hn);
  cvt(W_hh,  Whh_bf,  4 * Hn * Hn);
  cvt(out_w, outw_bf, Vn * Hn);
  gemm(enc_bf, Uaw_bf, Ua_b, UaK, Bn * Sn, Hn, Hn, 0);   // [8192,512]x[512,512]^T
  k_init<<<(Bn * Hn + 255) / 256, 256, 0, stream>>>(hidden, cell, type_idx, type_emb,
                                                    hbuf, cbuf, hbf, xbf, dec);

  // ---- 100 dependent decode steps ----
  for (int t = 0; t < Tn; ++t) {
    gemm(hbf, Waw_bf, Wa_b, qbuf, Bn, Hn, Hn, 0);                         // q
    k_attn_scores<<<(Bn * Sn) / 8, dim3(32, 8), 0, stream>>>(qbuf, UaK, Va_w,
                                                             Va_b, scores);
    k_softmax_ctx<<<Bn, Sn, 0, stream>>>(scores, enc, emb, dec, xbf, o_att, t);
    gemm(xbf, Wih_bf, b_ih, gates, Bn, 4 * Hn, 3 * Hn, 0);                // x W_ih^T
    gemm(hbf, Whh_bf, b_hh, gates, Bn, 4 * Hn, Hn, 1);                    // + h W_hh^T
    k_lstm<<<(Bn * Hn + 255) / 256, 256, 0, stream>>>(gates, cbuf, hbuf, hbf);
    gemm(hbf, outw_bf, out_b, logits, Bn, Vn, Hn, 0);                     // logits
    k_logsoftmax<<<Bn, 256, 0, stream>>>(logits, o_logp, dec, t);
  }
  k_final<<<(Bn * Hn + 255) / 256, 256, 0, stream>>>(hbuf, cbuf, o_h, o_c);
}
